// SalienceAttention_16578573763413
// MI455X (gfx1250) — matrix operand order repaired and checked by reading, not compiled, not run on hardware
//
#include <hip/hip_runtime.h>
#include <hip/hip_bf16.h>
#include <math.h>

// ---------------- problem constants ----------------
#define BB   32
#define CC   384
#define HH   56
#define WW   56
#define NN   (HH*WW)        // 3136
#define KK   98
#define NTOK 99
#define NHEAD 8
#define HD   48
#define MR   (BB*NTOK)      // 3168 token rows (multiple of 16)
#define BNT  (BB*NN)        // 100352 spatial tokens (multiple of 16)

// ---------------- WMMA types ----------------
typedef __bf16 bf16_t;
typedef bf16_t v16bf __attribute__((ext_vector_type(16)));
typedef float  v8f   __attribute__((ext_vector_type(8)));
typedef unsigned short u16x8 __attribute__((ext_vector_type(8)));

union Frag16 { v16bf v; u16x8 h[2]; };

__device__ __forceinline__ unsigned short f2bf(float f) {
  unsigned u = __float_as_uint(f);
  u += 0x7FFFu + ((u >> 16) & 1u);           // round-to-nearest-even
  return (unsigned short)(u >> 16);
}
__device__ __forceinline__ float bf2f(unsigned short h) {
  return __uint_as_float(((unsigned)h) << 16);
}

// =====================================================================
// Generic bf16 WMMA GEMM: out[row][col] = sum_k A[row][k]*W[col][k] + b
// A: (Mtotal x KDIM) bf16 row-major. W: (NOUT x KDIM) bf16 row-major.
// 256 threads = 8 waves; each wave computes a 16x16 tile; block covers
// 16 rows x 128 cols. MODE: 0=f32 out, 1=gelu->bf16, 2=+restored->bf16,
// 3=f32 NCHW scatter to d_out.
// =====================================================================
template<int KDIM, int NOUT, int MODE, int NSPATIAL>
__global__ __launch_bounds__(256)
void gemm_bf16_wmma(const unsigned short* __restrict__ A,
                    const unsigned short* __restrict__ Wb,
                    const float* __restrict__ bias,
                    float* __restrict__ outF,
                    unsigned short* __restrict__ outBF,
                    const float* __restrict__ attn,
                    const int* __restrict__ pos2tok) {
  const int tid  = threadIdx.x;
  const int wave = tid >> 5;
  const int lane = tid & 31;

  const int baseM = blockIdx.y * 16;
  const int baseN = (blockIdx.x * 8 + wave) * 16;

  const int rowA = baseM + (lane & 15);
  const int rowW = baseN + (lane & 15);
  const int offA = (lane >> 4) * 8;    // A: two 8-elem chunks (K .. K+7 / K+16..K+23)
  const int offB = (lane >> 4) * 8;    // B: two 8-elem chunks (K .. K+7 / K+16..K+23)

  const unsigned short* aptr = A  + (size_t)rowA * KDIM;
  const unsigned short* wptr = Wb + (size_t)rowW * KDIM;

  v8f acc = {0.f,0.f,0.f,0.f,0.f,0.f,0.f,0.f};

#pragma unroll
  for (int k = 0; k < KDIM; k += 32) {
    Frag16 a, b;
    a.h[0] = *(const u16x8*)(aptr + k + offA);
    a.h[1] = *(const u16x8*)(aptr + k + 16 + offA);
    b.h[0] = *(const u16x8*)(wptr + k + offB);
    b.h[1] = *(const u16x8*)(wptr + k + 16 + offB);
    acc = __builtin_amdgcn_wmma_f32_16x16x32_bf16(
        false, a.v, false, b.v, (short)0, acc, false, false);
  }

  const int col = baseN + (lane & 15);
  const float bv = bias[col];
#pragma unroll
  for (int e = 0; e < 8; e++) {
    const int row = baseM + e + (lane >> 4) * 8;
    float v = acc[e] + bv;
    if (MODE == 0) {
      outF[(size_t)row * NOUT + col] = v;
    } else if (MODE == 1) {
      // exact GELU: 0.5*x*(1+erf(x/sqrt(2)))
      float g = 0.5f * v * (1.0f + erff(v * 0.70710678118654752f));
      outBF[(size_t)row * NOUT + col] = f2bf(g);
    } else if (MODE == 2) {
      const int b_ = row / NSPATIAL;
      const int n  = row - b_ * NSPATIAL;
      const int t  = pos2tok[b_ * NSPATIAL + n];          // 0..97 top, 98 bg
      v += attn[((size_t)(b_ * NTOK + t)) * CC + col];    // restored
      outBF[(size_t)row * NOUT + col] = f2bf(v);
    } else { // MODE == 3: write NCHW f32
      const int b_ = row / NSPATIAL;
      const int n  = row - b_ * NSPATIAL;
      outF[((size_t)(b_ * NOUT + col)) * NSPATIAL + n] = v;
    }
  }
}

// ---------------- small helper kernels ----------------
__global__ void cvt_bf16_kernel(const float* __restrict__ src,
                                unsigned short* __restrict__ dst, int n) {
  int i = blockIdx.x * 256 + threadIdx.x;
  if (i < n) dst[i] = f2bf(src[i]);
}

// scores[b][n] = sum_c x[b][c][n]^2
__global__ void scores_kernel(const float* __restrict__ x,
                              float* __restrict__ scores) {
  int i = blockIdx.x * 256 + threadIdx.x;
  if (i >= BNT) return;
  int b = i / NN, n = i - b * NN;
  const float* p = x + (size_t)b * CC * NN + n;
  float s = 0.f;
  for (int c = 0; c < CC; c++) { float v = p[(size_t)c * NN]; s += v * v; }
  scores[i] = s;
}

// chsum[b][c] = sum_n x[b][c][n]   (one wave per row)
__global__ void chsum_kernel(const float* __restrict__ x,
                             float* __restrict__ chsum) {
  int row = blockIdx.x * (blockDim.x >> 5) + (threadIdx.x >> 5);
  if (row >= BB * CC) return;
  int lane = threadIdx.x & 31;
  const float* p = x + (size_t)row * NN;
  float s = 0.f;
  for (int n = lane; n < NN; n += 32) s += p[n];
  for (int off = 16; off > 0; off >>= 1) s += __shfl_xor(s, off, 32);
  if (lane == 0) chsum[row] = s;
}

// iterative top-98 argmax per batch (stable: ties -> lowest index)
__global__ __launch_bounds__(256) void topk_kernel(const float* __restrict__ scores,
                                                   int* __restrict__ top_idx) {
  int b = blockIdx.x;
  __shared__ float s[NN];
  __shared__ float rmax[256];
  __shared__ int   ridx[256];
  for (int i = threadIdx.x; i < NN; i += 256) s[i] = scores[b * NN + i];
  __syncthreads();
  for (int t = 0; t < KK; t++) {
    float best = -1e30f; int bi = 0x7fffffff;
    for (int i = threadIdx.x; i < NN; i += 256) {
      float v = s[i];
      if (v > best || (v == best && i < bi)) { best = v; bi = i; }
    }
    rmax[threadIdx.x] = best; ridx[threadIdx.x] = bi;
    __syncthreads();
    for (int off = 128; off > 0; off >>= 1) {
      if (threadIdx.x < off) {
        float ov = rmax[threadIdx.x + off]; int oi = ridx[threadIdx.x + off];
        if (ov > rmax[threadIdx.x] ||
            (ov == rmax[threadIdx.x] && oi < ridx[threadIdx.x])) {
          rmax[threadIdx.x] = ov; ridx[threadIdx.x] = oi;
        }
      }
      __syncthreads();
    }
    if (threadIdx.x == 0) { top_idx[b * KK + t] = ridx[0]; s[ridx[0]] = -1e30f; }
    __syncthreads();
  }
}

// tokens[b][t][c] (bf16) = gathered top tokens; token 98 = background mean
__global__ void gather_tokens_kernel(const float* __restrict__ x,
                                     const int* __restrict__ top_idx,
                                     const float* __restrict__ chsum,
                                     unsigned short* __restrict__ tokens) {
  int b = blockIdx.x;
  int c = threadIdx.x;                 // 384 threads
  const float* p = x + ((size_t)b * CC + c) * NN;
  float s = 0.f;
  for (int t = 0; t < KK; t++) {
    float v = p[top_idx[b * KK + t]];
    s += v;
    tokens[((size_t)(b * NTOK + t)) * CC + c] = f2bf(v);
  }
  float bg = (chsum[b * CC + c] - s) * (1.0f / (float)(NN - KK));
  tokens[((size_t)(b * NTOK + KK)) * CC + c] = f2bf(bg);
}

__global__ void pos2tok_init_kernel(int* __restrict__ p) {
  int i = blockIdx.x * 256 + threadIdx.x;
  if (i < BNT) p[i] = KK;              // default: background token
}
__global__ void pos2tok_scatter_kernel(const int* __restrict__ top_idx,
                                       int* __restrict__ p) {
  int i = blockIdx.x * 256 + threadIdx.x;
  if (i >= BB * KK) return;
  int b = i / KK, t = i - b * KK;
  p[b * NN + top_idx[i]] = t;
}

// per-(batch,head) attention over 99 tokens with online softmax
__global__ __launch_bounds__(128) void attention_kernel(const float* __restrict__ qkv,
                                                        float* __restrict__ attn_out) {
  int b = blockIdx.x, h = blockIdx.y;
  __shared__ float sq[NTOK * HD];
  __shared__ float sk[NTOK * HD];
  __shared__ float sv[NTOK * HD];
  for (int i = threadIdx.x; i < NTOK * HD; i += blockDim.x) {
    int t = i / HD, d = i - t * HD;
    size_t base = ((size_t)(b * NTOK + t)) * (3 * CC) + h * HD + d;
    sq[i] = qkv[base];
    sk[i] = qkv[base + CC];
    sv[i] = qkv[base + 2 * CC];
  }
  __syncthreads();
  const float scale = rsqrtf((float)HD);
  for (int i = threadIdx.x; i < NTOK; i += blockDim.x) {
    float m = -1e30f, l = 0.f;
    float acc[HD];
#pragma unroll
    for (int d = 0; d < HD; d++) acc[d] = 0.f;
    for (int j = 0; j < NTOK; j++) {
      float dot = 0.f;
#pragma unroll
      for (int d = 0; d < HD; d++) dot += sq[i * HD + d] * sk[j * HD + d];
      float s = dot * scale;
      float mn = fmaxf(m, s);
      float corr = __expf(m - mn);
      float p = __expf(s - mn);
      l = l * corr + p;
#pragma unroll
      for (int d = 0; d < HD; d++) acc[d] = acc[d] * corr + p * sv[j * HD + d];
      m = mn;
    }
    float inv = 1.0f / l;
#pragma unroll
    for (int d = 0; d < HD; d++)
      attn_out[((size_t)(b * NTOK + i)) * CC + h * HD + d] = acc[d] * inv;
  }
}

// per-channel mean/var over (B,H,W)
__global__ __launch_bounds__(256) void bn_stats_kernel(const float* __restrict__ x,
                                                       float* __restrict__ mean,
                                                       float* __restrict__ var) {
  int c = blockIdx.x;
  __shared__ float ssum[256], ssq[256];
  float s = 0.f, q = 0.f;
  for (int i = threadIdx.x; i < BNT; i += 256) {
    int b = i / NN, n = i - b * NN;
    float v = x[((size_t)b * CC + c) * NN + n];
    s += v; q += v * v;
  }
  ssum[threadIdx.x] = s; ssq[threadIdx.x] = q;
  __syncthreads();
  for (int off = 128; off > 0; off >>= 1) {
    if (threadIdx.x < off) {
      ssum[threadIdx.x] += ssum[threadIdx.x + off];
      ssq[threadIdx.x]  += ssq[threadIdx.x + off];
    }
    __syncthreads();
  }
  if (threadIdx.x == 0) {
    float mu = ssum[0] * (1.0f / (float)BNT);
    mean[c] = mu;
    var[c]  = ssq[0] * (1.0f / (float)BNT) - mu * mu;
  }
}

// xn (bf16, NHWC layout: (b*N+n)*C + c) = BN(x)
__global__ void bn_apply_kernel(const float* __restrict__ x,
                                const float* __restrict__ mean,
                                const float* __restrict__ var,
                                const float* __restrict__ g,
                                const float* __restrict__ bt,
                                unsigned short* __restrict__ xn) {
  size_t i = (size_t)blockIdx.x * 256 + threadIdx.x;   // (b,n,c), c fastest
  if (i >= (size_t)BNT * CC) return;
  int c = (int)(i % CC);
  size_t pos = i / CC;
  int b = (int)(pos / NN), n = (int)(pos - (size_t)b * NN);
  float v = x[((size_t)b * CC + c) * NN + n];
  float r = (v - mean[c]) * rsqrtf(var[c] + 1e-5f) * g[c] + bt[c];
  xn[i] = f2bf(r);
}

// depthwise 3x3 SAME on NHWC bf16, + bias
__global__ void dwconv_kernel(const unsigned short* __restrict__ h1,
                              const float* __restrict__ dww,
                              const float* __restrict__ dwb,
                              unsigned short* __restrict__ h2) {
  size_t i = (size_t)blockIdx.x * 256 + threadIdx.x;   // (b,y,x,c), c fastest
  if (i >= (size_t)BNT * CC) return;
  int c = (int)(i % CC);
  size_t pos = i / CC;
  int xx = (int)(pos % WW);
  size_t p2 = pos / WW;
  int y = (int)(p2 % HH);
  int b = (int)(p2 / HH);
  float acc = dwb[c];
#pragma unroll
  for (int dy = -1; dy <= 1; dy++) {
#pragma unroll
    for (int dx = -1; dx <= 1; dx++) {
      int yy = y + dy, x2 = xx + dx;
      if (yy >= 0 && yy < HH && x2 >= 0 && x2 < WW) {
        float v = bf2f(h1[(((size_t)(b * HH + yy)) * WW + x2) * CC + c]);
        acc += v * dww[c * 9 + (dy + 1) * 3 + (dx + 1)];
      }
    }
  }
  h2[i] = f2bf(acc);
}

// =====================================================================
extern "C" void kernel_launch(void* const* d_in, const int* in_sizes, int n_in,
                              void* d_out, int out_size, void* d_ws, size_t ws_size,
                              hipStream_t stream) {
  const float* x      = (const float*)d_in[0];
  const float* qkv_w  = (const float*)d_in[1];
  const float* qkv_b  = (const float*)d_in[2];
  const float* proj_w = (const float*)d_in[3];
  const float* proj_b = (const float*)d_in[4];
  const float* bn_g   = (const float*)d_in[5];
  const float* bn_b   = (const float*)d_in[6];
  const float* fc1_w  = (const float*)d_in[7];
  const float* fc1_b  = (const float*)d_in[8];
  const float* fc2_w  = (const float*)d_in[9];
  const float* fc2_b  = (const float*)d_in[10];
  const float* dw_w   = (const float*)d_in[11];
  const float* dw_b   = (const float*)d_in[12];
  float* out = (float*)d_out;

  // -------- workspace carve-out --------
  char* ws = (char*)d_ws;
  size_t off = 0;
  auto alloc = [&](size_t bytes) -> void* {
    void* p = ws + off;
    off = (off + bytes + 255) & ~(size_t)255;
    return p;
  };
  float*          scores = (float*)alloc((size_t)BNT * 4);
  float*          chsum  = (float*)alloc((size_t)BB * CC * 4);
  int*            topidx = (int*)alloc((size_t)BB * KK * 4);
  int*            p2t    = (int*)alloc((size_t)BNT * 4);
  float*          mean   = (float*)alloc(CC * 4);
  float*          var    = (float*)alloc(CC * 4);
  unsigned short* wq     = (unsigned short*)alloc((size_t)3 * CC * CC * 2);
  unsigned short* wf1    = (unsigned short*)alloc((size_t)CC * CC * 2);
  unsigned short* wf2    = (unsigned short*)alloc((size_t)CC * CC * 2);
  unsigned short* wp     = (unsigned short*)alloc((size_t)CC * CC * 2);
  unsigned short* tokens = (unsigned short*)alloc((size_t)MR * CC * 2);
  float*          qkvo   = (float*)alloc((size_t)MR * 3 * CC * 4);
  float*          attn   = (float*)alloc((size_t)MR * CC * 4);
  unsigned short* bufA   = (unsigned short*)alloc((size_t)BNT * CC * 2);
  unsigned short* bufB   = (unsigned short*)alloc((size_t)BNT * CC * 2);

  // -------- weight conversion to bf16 --------
  cvt_bf16_kernel<<<(3 * CC * CC + 255) / 256, 256, 0, stream>>>(qkv_w, wq, 3 * CC * CC);
  cvt_bf16_kernel<<<(CC * CC + 255) / 256, 256, 0, stream>>>(fc1_w, wf1, CC * CC);
  cvt_bf16_kernel<<<(CC * CC + 255) / 256, 256, 0, stream>>>(fc2_w, wf2, CC * CC);
  cvt_bf16_kernel<<<(CC * CC + 255) / 256, 256, 0, stream>>>(proj_w, wp, CC * CC);

  // -------- salience path --------
  scores_kernel<<<(BNT + 255) / 256, 256, 0, stream>>>(x, scores);
  chsum_kernel<<<(BB * CC + 7) / 8, 256, 0, stream>>>(x, chsum);
  topk_kernel<<<BB, 256, 0, stream>>>(scores, topidx);
  gather_tokens_kernel<<<BB, CC, 0, stream>>>(x, topidx, chsum, tokens);
  pos2tok_init_kernel<<<(BNT + 255) / 256, 256, 0, stream>>>(p2t);
  pos2tok_scatter_kernel<<<(BB * KK + 255) / 256, 256, 0, stream>>>(topidx, p2t);

  // QKV GEMM: (3168 x 384) @ (384 x 1152) -> f32
  gemm_bf16_wmma<CC, 3 * CC, 0, 1><<<dim3((3 * CC) / 128, MR / 16), 256, 0, stream>>>(
      tokens, wq, qkv_b, qkvo, nullptr, nullptr, nullptr);
  attention_kernel<<<dim3(BB, NHEAD), 128, 0, stream>>>(qkvo, attn);

  // -------- conv/MLP path --------
  bn_stats_kernel<<<CC, 256, 0, stream>>>(x, mean, var);
  const int ELT_BLKS = (int)(((size_t)BNT * CC + 255) / 256);
  bn_apply_kernel<<<ELT_BLKS, 256, 0, stream>>>(x, mean, var, bn_g, bn_b, bufA);

  // fc1 + GELU: (100352 x 384) @ (384 x 384) -> bf16
  gemm_bf16_wmma<CC, CC, 1, NN><<<dim3(CC / 128, BNT / 16), 256, 0, stream>>>(
      bufA, wf1, fc1_b, nullptr, bufB, nullptr, nullptr);

  // depthwise 3x3 (NHWC): bufB -> bufA
  dwconv_kernel<<<ELT_BLKS, 256, 0, stream>>>(bufB, dw_w, dw_b, bufA);

  // fc2 + restored-attention fuse: bufA -> bufB (combined)
  gemm_bf16_wmma<CC, CC, 2, NN><<<dim3(CC / 128, BNT / 16), 256, 0, stream>>>(
      bufA, wf2, fc2_b, nullptr, bufB, attn, p2t);

  // proj: bufB -> d_out (NCHW f32)
  gemm_bf16_wmma<CC, CC, 3, NN><<<dim3(CC / 128, BNT / 16), 256, 0, stream>>>(
      bufB, wp, proj_b, out, nullptr, nullptr, nullptr);
}
